// MultiHeadAttention_15668040696499
// MI455X (gfx1250) — matrix operand order repaired and checked by reading, not compiled, not run on hardware
//
#include <hip/hip_runtime.h>
#include <hip/hip_bf16.h>
#include <stdint.h>

// ---------------------------------------------------------------------------
// MI455X (gfx1250) multi-head attention forward.
// All heavy math on v_wmma_f32_16x16x32_bf16 (f32 accumulate).
// Softmax reductions on v_permlane16_b32 (VALU, co-executes with WMMA).
// bf16 GEMM A-tiles staged with GLOBAL_LOAD_ASYNC_TO_LDS_B128 when available.
// ---------------------------------------------------------------------------

typedef __attribute__((ext_vector_type(16))) __bf16   v16bf;
typedef __attribute__((ext_vector_type(8)))  float    v8f;
typedef __attribute__((ext_vector_type(4)))  uint32_t u32x4;
typedef __attribute__((ext_vector_type(2)))  uint32_t u32x2;
typedef __attribute__((ext_vector_type(4)))  float    f32x4;
typedef __attribute__((ext_vector_type(4)))  int      i32x4;

#define DIMN 1024
#define SEQ  2048
#define NH   16
#define HD   64

#if defined(__has_builtin)
#  if __has_builtin(__builtin_amdgcn_permlane16)
#    define HAVE_PERMLANE16 1
#  endif
#  if __has_builtin(__builtin_amdgcn_global_load_async_to_lds_b128)
#    define HAVE_ASYNC_LDS 1
#  endif
#endif

__device__ __forceinline__ uint16_t f2bf(float f) {
  uint32_t u = __builtin_bit_cast(uint32_t, f);
  uint32_t r = u + 0x7FFFu + ((u >> 16) & 1u);   // round-to-nearest-even
  return (uint16_t)(r >> 16);
}

struct U32x8 { u32x4 lo, hi; };
__device__ __forceinline__ v16bf make_frag(u32x4 lo, u32x4 hi) {
  U32x8 t; t.lo = lo; t.hi = hi;
  return __builtin_bit_cast(v16bf, t);
}
__device__ __forceinline__ v8f v8f_zero() {
  v8f z = {0.f,0.f,0.f,0.f,0.f,0.f,0.f,0.f};
  return z;
}

#ifdef HAVE_ASYNC_LDS
// ASYNCcnt-tracked DMA: 16 bytes global -> LDS, no VGPR staging.
__device__ __forceinline__ void async_copy16(const void* gsrc, void* ldst) {
  __builtin_amdgcn_global_load_async_to_lds_b128(
      (__attribute__((address_space(1))) i32x4*)(void*)gsrc,
      (__attribute__((address_space(3))) i32x4*)ldst,
      0, 0);
}
#endif

// ---- 16-lane butterfly reductions via v_permlane16_b32 --------------------
#ifdef HAVE_PERMLANE16
__device__ __forceinline__ float lane16_xor(float x, unsigned lo, unsigned hi) {
  unsigned i = __builtin_bit_cast(unsigned, x);
  unsigned r = __builtin_amdgcn_permlane16(i, i, lo, hi, false, false);
  return __builtin_bit_cast(float, r);
}
#endif

__device__ __forceinline__ float rowmax16(float x) {
#ifdef HAVE_PERMLANE16
  x = fmaxf(x, lane16_xor(x, 0x67452301u, 0xEFCDAB89u));  // xor 1
  x = fmaxf(x, lane16_xor(x, 0x54761032u, 0xDCFE98BAu));  // xor 2
  x = fmaxf(x, lane16_xor(x, 0x32107654u, 0xBA98FEDCu));  // xor 4
  x = fmaxf(x, lane16_xor(x, 0xFEDCBA98u, 0x76543210u));  // xor 8
#else
  #pragma unroll
  for (int m = 1; m < 16; m <<= 1) x = fmaxf(x, __shfl_xor(x, m, 32));
#endif
  return x;
}
__device__ __forceinline__ float rowsum16(float x) {
#ifdef HAVE_PERMLANE16
  x += lane16_xor(x, 0x67452301u, 0xEFCDAB89u);
  x += lane16_xor(x, 0x54761032u, 0xDCFE98BAu);
  x += lane16_xor(x, 0x32107654u, 0xBA98FEDCu);
  x += lane16_xor(x, 0xFEDCBA98u, 0x76543210u);
#else
  #pragma unroll
  for (int m = 1; m < 16; m <<= 1) x += __shfl_xor(x, m, 32);
#endif
  return x;
}

// ---------------------------------------------------------------------------
// GEMM: Out[M x DIMN] = A[M x DIMN] * W[DIMN x DIMN] + bias
// A_F32: A is f32 (converted to bf16 while staging) else bf16.
// OMODE 0: bf16 out, [b][h][s][d]   (Q,K projections)
// OMODE 1: bf16 out, [b][h][d][s]   (V projection, pre-transposed)
// OMODE 2: f32 out, row-major [m][n] (final projection -> d_out)
// ---------------------------------------------------------------------------
template<bool A_F32, int OMODE>
__global__ __launch_bounds__(256)
void gemm_kernel(const void* __restrict__ Aptr, const float* __restrict__ W,
                 const float* __restrict__ bias, void* __restrict__ Out, int M)
{
  constexpr int BM = 128, BN = 128, BK = 32;
  constexpr int LDA = 40;                       // ushorts per row = 80B pitch
  __shared__ uint16_t lA[BM * LDA];             // A tile, [m][k] bf16
  __shared__ uint16_t lB[BN * LDA];             // B^T tile, [n][k] bf16

  const int tid  = threadIdx.x;
  const int lane = tid & 31, l16 = lane & 15, hi = lane >> 4;
  const int w    = tid >> 5;                    // 8 waves
  const int wm   = w >> 1, wn = w & 1;          // 4 x 2 wave grid
  const int m0   = blockIdx.y * BM;
  const int n0   = blockIdx.x * BN;
  const int m0w  = wm * 32, n0w = wn * 64;

  v8f acc[2][4];
  #pragma unroll
  for (int i = 0; i < 2; ++i)
    #pragma unroll
    for (int j = 0; j < 4; ++j) acc[i][j] = v8f_zero();

  for (int kk = 0; kk < DIMN; kk += BK) {
    // ---- stage A tile ----
    if constexpr (A_F32) {
      const float* A = (const float*)Aptr;
      #pragma unroll
      for (int i = 0; i < 4; ++i) {
        int idx = tid + i * 256;                // 1024 float4 chunks
        int row = idx >> 3, k4 = idx & 7;
        f32x4 v = *(const f32x4*)(A + (size_t)(m0 + row) * DIMN + kk + k4 * 4);
        uint32_t p0 = (uint32_t)f2bf(v.x) | ((uint32_t)f2bf(v.y) << 16);
        uint32_t p1 = (uint32_t)f2bf(v.z) | ((uint32_t)f2bf(v.w) << 16);
        u32x2 pk; pk.x = p0; pk.y = p1;
        *(u32x2*)&lA[row * LDA + k4 * 4] = pk;
      }
    } else {
      const uint16_t* A = (const uint16_t*)Aptr;
      #pragma unroll
      for (int i = 0; i < 2; ++i) {
        int idx = tid + i * 256;                // 512 8-bf16 (16B) chunks
        int row = idx >> 2, k8 = idx & 3;
        const uint16_t* gsrc = A + (size_t)(m0 + row) * DIMN + kk + k8 * 8;
        uint16_t*       ldst = &lA[row * LDA + k8 * 8];
#ifdef HAVE_ASYNC_LDS
        async_copy16(gsrc, ldst);
#else
        *(u32x4*)ldst = *(const u32x4*)gsrc;
#endif
      }
    }
    // ---- stage B^T tile (W is [k][n] row-major f32) ----
    #pragma unroll
    for (int i = 0; i < 4; ++i) {
      int idx = tid + i * 256;                  // 1024 float4 chunks
      int kr = idx >> 5, n4 = idx & 31;
      const float* wsrc = W + (size_t)(kk + kr) * DIMN + n0 + n4 * 4;
      f32x4 v = *(const f32x4*)wsrc;
      if (kk + BK < DIMN)                       // prefetch next W tile
        __builtin_prefetch(wsrc + (size_t)BK * DIMN, 0, 1);
      lB[(n4 * 4 + 0) * LDA + kr] = f2bf(v.x);
      lB[(n4 * 4 + 1) * LDA + kr] = f2bf(v.y);
      lB[(n4 * 4 + 2) * LDA + kr] = f2bf(v.z);
      lB[(n4 * 4 + 3) * LDA + kr] = f2bf(v.w);
    }
#ifdef HAVE_ASYNC_LDS
    if constexpr (!A_F32)
      asm volatile("s_wait_asynccnt 0x0" ::: "memory");
#endif
    __syncthreads();

    // ---- gather fragments (ISA 16-bit A / B^T VGPR layouts) ----
    v16bf af[2], bfr[4];
    #pragma unroll
    for (int sm = 0; sm < 2; ++sm) {
      int base = (m0w + sm * 16 + l16) * LDA + hi * 8;   // lane-lo: K0..7 / lane-hi: K8..15
      af[sm] = make_frag(*(const u32x4*)&lA[base], *(const u32x4*)&lA[base + 16]);
    }
    #pragma unroll
    for (int sn = 0; sn < 4; ++sn) {
      int base = (n0w + sn * 16 + l16) * LDA + hi * 8;   // lane-lo: K0..7 / lane-hi: K8..15
      bfr[sn] = make_frag(*(const u32x4*)&lB[base], *(const u32x4*)&lB[base + 16]);
    }
    #pragma unroll
    for (int sm = 0; sm < 2; ++sm)
      #pragma unroll
      for (int sn = 0; sn < 4; ++sn)
        acc[sm][sn] = __builtin_amdgcn_wmma_f32_16x16x32_bf16(
            false, af[sm], false, bfr[sn], (short)0, acc[sm][sn], false, false);
    __syncthreads();
  }

  // ---- epilogue: bias + layout-specific store ----
  #pragma unroll
  for (int sm = 0; sm < 2; ++sm)
    #pragma unroll
    for (int sn = 0; sn < 4; ++sn) {
      int n = n0 + n0w + sn * 16 + l16;
      float bv = bias[n];
      #pragma unroll
      for (int r = 0; r < 8; ++r) {
        int m = m0 + m0w + sm * 16 + hi * 8 + r;   // C layout: VGPR r, lane-half -> row
        float val = acc[sm][sn][r] + bv;
        if constexpr (OMODE == 2) {
          ((float*)Out)[(size_t)m * DIMN + n] = val;
        } else {
          int b = m / SEQ, s = m % SEQ;
          int h = n / HD,  d = n % HD;
          if constexpr (OMODE == 0)
            ((uint16_t*)Out)[(((size_t)(b * NH + h)) * SEQ + s) * HD + d] = f2bf(val);
          else
            ((uint16_t*)Out)[(((size_t)(b * NH + h)) * HD + d) * SEQ + s] = f2bf(val);
        }
      }
    }
}

// ---------------------------------------------------------------------------
// Flash attention: one wave owns 16 query rows; keys processed 32 at a time.
// q,k in [b][h][s][d] bf16; v in [b][h][d][s] bf16; out in [b][s][h][d] bf16.
// ---------------------------------------------------------------------------
__global__ __launch_bounds__(256)
void attn_kernel(const uint16_t* __restrict__ qp, const uint16_t* __restrict__ kp,
                 const uint16_t* __restrict__ vt, uint16_t* __restrict__ outp)
{
  constexpr float SCALE = 0.125f;               // 1/sqrt(64)
  constexpr float LOG2E = 1.44269504088896340736f;
  __shared__ uint16_t lP[8][16 * 40];           // per-wave P re-swizzle tile

  const int tid  = threadIdx.x;
  const int lane = tid & 31, l16 = lane & 15, hi = lane >> 4;
  const int w    = tid >> 5;
  const int bh   = blockIdx.y;                  // b*NH + h
  const int b    = bh >> 4, h = bh & 15;
  const int q0   = blockIdx.x * 128 + w * 16;

  const uint16_t* qb = qp + (size_t)bh * SEQ * HD;
  const uint16_t* kb = kp + (size_t)bh * SEQ * HD;
  const uint16_t* vb = vt + (size_t)bh * HD * SEQ;

  // Q A-fragments (d 0..31 and 32..63), straight from global in A layout
  v16bf aq[2];
  {
    const uint16_t* qrow = qb + (size_t)(q0 + l16) * HD;
    #pragma unroll
    for (int j = 0; j < 2; ++j) {
      const uint16_t* p = qrow + j * 32 + hi * 8;
      aq[j] = make_frag(*(const u32x4*)p, *(const u32x4*)(p + 16));
    }
  }

  v8f o[4];
  float mrow[8], lrow[8];
  #pragma unroll
  for (int j = 0; j < 4; ++j) o[j] = v8f_zero();
  #pragma unroll
  for (int r = 0; r < 8; ++r) { mrow[r] = -1e30f; lrow[r] = 0.f; }

  for (int sk = 0; sk < SEQ; sk += 32) {
    // ---- scores: S[16q x 32k] via 4 WMMAs (K frags straight from global) ----
    v8f sc[2];
    #pragma unroll
    for (int nh2 = 0; nh2 < 2; ++nh2) {
      const uint16_t* krow = kb + (size_t)(sk + nh2 * 16 + l16) * HD;
      v8f t = v8f_zero();
      #pragma unroll
      for (int j = 0; j < 2; ++j) {
        const uint16_t* p = krow + j * 32 + hi * 8;
        v16bf kf = make_frag(*(const u32x4*)p, *(const u32x4*)(p + 16));
        t = __builtin_amdgcn_wmma_f32_16x16x32_bf16(
                false, aq[j], false, kf, (short)0, t, false, false);
      }
      sc[nh2] = t;
    }

    // ---- online softmax (rows live on accumulator reg r + lane-half) ----
    #pragma unroll
    for (int r = 0; r < 8; ++r) {
      float s0 = sc[0][r] * SCALE, s1 = sc[1][r] * SCALE;
      float t  = rowmax16(fmaxf(s0, s1));
      float mn    = fmaxf(mrow[r], t);
      float alpha = exp2f((mrow[r] - mn) * LOG2E);
      float e0    = exp2f((s0 - mn) * LOG2E);
      float e1    = exp2f((s1 - mn) * LOG2E);
      float rs    = rowsum16(e0 + e1);
      lrow[r] = lrow[r] * alpha + rs;
      mrow[r] = mn;
      #pragma unroll
      for (int j = 0; j < 4; ++j) o[j][r] *= alpha;
      // scatter P into per-wave LDS tile ([m][k], 80B pitch)
      int mr = (r + hi * 8) * 40;
      lP[w][mr + l16]      = f2bf(e0);
      lP[w][mr + 16 + l16] = f2bf(e1);
    }

    // ---- P A-fragment from LDS (same-wave DS ops are in-order) ----
    v16bf ap;
    {
      const uint16_t* base = &lP[w][l16 * 40 + hi * 8];
      ap = make_frag(*(const u32x4*)base, *(const u32x4*)(base + 16));
    }

    // ---- O += P * V  (V^T frags straight from global) ----
    #pragma unroll
    for (int j = 0; j < 4; ++j) {
      const uint16_t* vrow = vb + (size_t)(j * 16 + l16) * SEQ + sk + hi * 8;
      v16bf vf = make_frag(*(const u32x4*)vrow, *(const u32x4*)(vrow + 16));
      o[j] = __builtin_amdgcn_wmma_f32_16x16x32_bf16(
                 false, ap, false, vf, (short)0, o[j], false, false);
    }
  }

  // ---- normalize and store [b][s][h][d] bf16 ----
  #pragma unroll
  for (int r = 0; r < 8; ++r) {
    float inv = 1.0f / lrow[r];
    int s = q0 + r + hi * 8;
    size_t base = ((size_t)(b * SEQ + s) * NH + h) * HD;
    #pragma unroll
    for (int j = 0; j < 4; ++j)
      outp[base + j * 16 + l16] = f2bf(o[j][r] * inv);
  }
}

// ---------------------------------------------------------------------------
extern "C" void kernel_launch(void* const* d_in, const int* in_sizes, int n_in,
                              void* d_out, int out_size, void* d_ws, size_t ws_size,
                              hipStream_t stream) {
  const float* Q  = (const float*)d_in[0];
  const float* K  = (const float*)d_in[1];
  const float* V  = (const float*)d_in[2];
  const float* Wq = (const float*)d_in[3];
  const float* bq = (const float*)d_in[4];
  const float* Wk = (const float*)d_in[5];
  const float* bk = (const float*)d_in[6];
  const float* Wv = (const float*)d_in[7];
  const float* bv = (const float*)d_in[8];
  const float* Wo = (const float*)d_in[9];
  const float* bo = (const float*)d_in[10];

  const int M = in_sizes[0] / DIMN;             // 8192
  const int B = M / SEQ;                        // 4

  char* ws = (char*)d_ws;
  const size_t seg = (size_t)B * NH * SEQ * HD * sizeof(uint16_t); // 16 MB
  uint16_t* qproj = (uint16_t*)(ws);
  uint16_t* kproj = (uint16_t*)(ws + seg);
  uint16_t* vtra  = (uint16_t*)(ws + 2 * seg);
  uint16_t* attn  = (uint16_t*)(ws + 3 * seg);

  dim3 gg(DIMN / 128, M / 128), bb(256, 1, 1);
  gemm_kernel<true, 0><<<gg, bb, 0, stream>>>(Q, Wq, bq, qproj, M);
  gemm_kernel<true, 0><<<gg, bb, 0, stream>>>(K, Wk, bk, kproj, M);
  gemm_kernel<true, 1><<<gg, bb, 0, stream>>>(V, Wv, bv, vtra, M);

  dim3 ga(SEQ / 128, B * NH);
  attn_kernel<<<ga, bb, 0, stream>>>(qproj, kproj, vtra, attn);

  gemm_kernel<false, 2><<<gg, bb, 0, stream>>>(attn, Wo, bo, d_out, M);
}
